// CrfRnnLayerForInstance_56367150793149
// MI455X (gfx1250) — compile-verified, matrix operand order checked by reading, not run on hardware
//
#include <hip/hip_runtime.h>
#include <hip/hip_bf16.h>

typedef __attribute__((ext_vector_type(16))) _Float16 v16h;
typedef __attribute__((ext_vector_type(8)))  float    v8f;

#define Hh   512
#define Ww   512
#define Dd   16
#define HW   (Hh * Ww)          // 262144 = 2^18
#define HW16 (HW * Dd)          // 4194304
#define RADS 9                  // spatial radius (19 taps)
#define RADB 2                  // bilateral radius (5x5)
#define INV2A (1.0f / (2.0f * 160.0f * 160.0f))
#define INV2B (1.0f / (2.0f * 3.0f * 3.0f))

// ---------------------------------------------------------------------------
// Small one-block kernel: fused weight matrix A = [-(C*Ws) | -(C*Wb)] in f16,
// plus the 19-tap Gaussian table.
// ---------------------------------------------------------------------------
__global__ void k_init_small(const float* __restrict__ Ws,
                             const float* __restrict__ Wb,
                             const float* __restrict__ Cm,
                             _Float16* __restrict__ A,
                             float* __restrict__ GK) {
  int tid = threadIdx.x;
  if (tid < 512) {
    int i = tid >> 5;          // output row (0..15)
    int j = tid & 31;          // fused column (0..31)
    float s = 0.f;
    if (j < 16) {
      for (int k = 0; k < 16; ++k) s += Cm[i * 16 + k] * Ws[k * 16 + j];
    } else {
      int jj = j - 16;
      for (int k = 0; k < 16; ++k) s += Cm[i * 16 + k] * Wb[k * 16 + jj];
    }
    A[i * 32 + j] = (_Float16)(-s);
  } else if (tid < 512 + 2 * RADS + 1) {
    int t = tid - 512;
    int d = t - RADS;
    GK[t] = __expf(-(float)(d * d) * (1.0f / 18.0f));
  }
}

// ---------------------------------------------------------------------------
// unaries = box*bw + glob*gw ; transpose plane-major -> pixel-major.
// Writes both U (kept) and Q (iteration state).
// ---------------------------------------------------------------------------
__global__ void k_prep(const float* __restrict__ box,
                       const float* __restrict__ glob,
                       const float* __restrict__ bwp,
                       const float* __restrict__ gwp,
                       float* __restrict__ U, float* __restrict__ Q) {
  int idx = blockIdx.x * blockDim.x + threadIdx.x;   // d*HW + pix
  if (idx >= HW16) return;
  int d   = idx >> 18;
  int pix = idx & (HW - 1);
  float u = box[idx] * bwp[0] + glob[idx] * gwp[0];
  U[pix * 16 + d] = u;
  Q[pix * 16 + d] = u;
}

// ---------------------------------------------------------------------------
// One-time: 25 bilateral weight maps, reciprocal bilateral norm, reciprocal
// spatial norm (separable product s(y)*s(x)).
// ---------------------------------------------------------------------------
__global__ void k_bw(const float* __restrict__ img,    // (H,W,3)
                     const float* __restrict__ GK,
                     float* __restrict__ W25,          // [25][HW]
                     float* __restrict__ RBN,          // [HW]
                     float* __restrict__ RSN) {        // [HW]
  int pix = blockIdx.x * blockDim.x + threadIdx.x;
  if (pix >= HW) return;
  int x = pix & (Ww - 1);
  int y = pix >> 9;

  float kk[2 * RADS + 1];
  #pragma unroll
  for (int t = 0; t < 2 * RADS + 1; ++t) kk[t] = GK[t];

  float sy = 0.f, sx = 0.f;
  #pragma unroll
  for (int t = 0; t < 2 * RADS + 1; ++t) {
    int yy = y + t - RADS; if (yy >= 0 && yy < Hh) sy += kk[t];
    int xx = x + t - RADS; if (xx >= 0 && xx < Ww) sx += kk[t];
  }
  RSN[pix] = 1.0f / (sy * sx);

  float r0 = img[pix * 3 + 0], g0 = img[pix * 3 + 1], b0 = img[pix * 3 + 2];
  float norm = 0.f;
  #pragma unroll
  for (int dy = -RADB; dy <= RADB; ++dy) {
    #pragma unroll
    for (int dx = -RADB; dx <= RADB; ++dx) {
      int t = (dy + RADB) * 5 + (dx + RADB);
      float w = 0.f;
      int ny = y + dy, nx = x + dx;
      if (ny >= 0 && ny < Hh && nx >= 0 && nx < Ww) {
        int np = ny * Ww + nx;
        float dr = r0 - img[np * 3 + 0];
        float dg = g0 - img[np * 3 + 1];
        float db = b0 - img[np * 3 + 2];
        w = __expf(-(float)(dy * dy + dx * dx) * INV2A
                   - (dr * dr + dg * dg + db * db) * INV2B);
      }
      W25[t * HW + pix] = w;
      norm += w;
    }
  }
  RBN[pix] = 1.0f / norm;
}

// ---------------------------------------------------------------------------
// Softmax over the 16 contiguous channels per pixel.
// ---------------------------------------------------------------------------
__global__ void k_softmax(const float* __restrict__ Q, float* __restrict__ P) {
  int pix = blockIdx.x * blockDim.x + threadIdx.x;
  if (pix >= HW) return;
  const float* qp = Q + pix * 16;
  float v[16];
  #pragma unroll
  for (int i = 0; i < 16; ++i) v[i] = qp[i];
  float m = v[0];
  #pragma unroll
  for (int i = 1; i < 16; ++i) m = fmaxf(m, v[i]);
  float s = 0.f;
  #pragma unroll
  for (int i = 0; i < 16; ++i) { v[i] = __expf(v[i] - m); s += v[i]; }
  float r = 1.0f / s;
  float* pp = P + pix * 16;
  #pragma unroll
  for (int i = 0; i < 16; ++i) pp[i] = v[i] * r;
}

// ---------------------------------------------------------------------------
// Separable spatial filter, vertical then horizontal pass.
// Thread handles a float4 (4 channels) of one pixel.
// ---------------------------------------------------------------------------
__global__ void k_spat_v(const float* __restrict__ P, float* __restrict__ T,
                         const float* __restrict__ GK) {
  int idx = blockIdx.x * blockDim.x + threadIdx.x;   // pix*4 + c4
  if (idx >= HW * 4) return;
  int c4  = idx & 3;
  int pix = idx >> 2;
  int x = pix & (Ww - 1);
  int y = pix >> 9;
  float a0 = 0.f, a1 = 0.f, a2 = 0.f, a3 = 0.f;
  for (int t = 0; t < 2 * RADS + 1; ++t) {
    int yy = y + t - RADS;
    if (yy >= 0 && yy < Hh) {
      float k = GK[t];
      const float* p = P + ((yy * Ww + x) * 16 + c4 * 4);
      a0 += k * p[0]; a1 += k * p[1]; a2 += k * p[2]; a3 += k * p[3];
    }
  }
  float* o = T + (pix * 16 + c4 * 4);
  o[0] = a0; o[1] = a1; o[2] = a2; o[3] = a3;
}

__global__ void k_spat_h(const float* __restrict__ T, float* __restrict__ SP,
                         const float* __restrict__ GK,
                         const float* __restrict__ RSN) {
  int idx = blockIdx.x * blockDim.x + threadIdx.x;
  if (idx >= HW * 4) return;
  int c4  = idx & 3;
  int pix = idx >> 2;
  int x = pix & (Ww - 1);
  int y = pix >> 9;
  float a0 = 0.f, a1 = 0.f, a2 = 0.f, a3 = 0.f;
  for (int t = 0; t < 2 * RADS + 1; ++t) {
    int xx = x + t - RADS;
    if (xx >= 0 && xx < Ww) {
      float k = GK[t];
      const float* p = T + ((y * Ww + xx) * 16 + c4 * 4);
      a0 += k * p[0]; a1 += k * p[1]; a2 += k * p[2]; a3 += k * p[3];
    }
  }
  float rn = RSN[pix];
  float* o = SP + (pix * 16 + c4 * 4);
  o[0] = a0 * rn; o[1] = a1 * rn; o[2] = a2 * rn; o[3] = a3 * rn;
}

// ---------------------------------------------------------------------------
// Bilateral apply: precomputed 25 weight maps * p, via LDS tile with halo.
// Block = 16x16 pixels; LDS = 20x20 pixels x 16 channels = 25.6 KB.
// ---------------------------------------------------------------------------
#define TILE 16
#define HALO 2
#define LW   (TILE + 2 * HALO)     // 20

__global__ void k_bilateral(const float* __restrict__ P,
                            const float* __restrict__ W25,
                            const float* __restrict__ RBN,
                            float* __restrict__ BL) {
  __shared__ float lds[LW * LW * 16];
  int tx = threadIdx.x, ty = threadIdx.y;
  int gx0 = blockIdx.x * TILE, gy0 = blockIdx.y * TILE;
  int tflat = ty * TILE + tx;                  // 0..255

  // cooperative staging: LW*LW slots, 4 float4 each = 1600 float4 chunks
  for (int chunk = tflat; chunk < LW * LW * 4; chunk += TILE * TILE) {
    int slot = chunk >> 2;
    int c4   = chunk & 3;
    int ly = slot / LW, lx = slot % LW;
    int gy = gy0 - HALO + ly, gx = gx0 - HALO + lx;
    float v0 = 0.f, v1 = 0.f, v2 = 0.f, v3 = 0.f;
    if (gy >= 0 && gy < Hh && gx >= 0 && gx < Ww) {
      const float* p = P + ((gy * Ww + gx) * 16 + c4 * 4);
      v0 = p[0]; v1 = p[1]; v2 = p[2]; v3 = p[3];
    }
    float* l = lds + (slot * 16 + c4 * 4);
    l[0] = v0; l[1] = v1; l[2] = v2; l[3] = v3;
  }
  __syncthreads();

  int x = gx0 + tx, y = gy0 + ty;
  int pix = y * Ww + x;
  float a[16];
  #pragma unroll
  for (int i = 0; i < 16; ++i) a[i] = 0.f;

  #pragma unroll
  for (int dy = -RADB; dy <= RADB; ++dy) {
    #pragma unroll
    for (int dx = -RADB; dx <= RADB; ++dx) {
      int t = (dy + RADB) * 5 + (dx + RADB);
      float w = W25[t * HW + pix];
      const float* l = lds + (((ty + HALO + dy) * LW + (tx + HALO + dx)) * 16);
      #pragma unroll
      for (int i = 0; i < 16; ++i) a[i] += w * l[i];
    }
  }
  float rn = RBN[pix];
  float* o = BL + pix * 16;
  #pragma unroll
  for (int i = 0; i < 16; ++i) o[i] = a[i] * rn;
}

// ---------------------------------------------------------------------------
// Message + compatibility + update, one WMMA per 16-pixel tile:
//   q_tile(16x16) = A(16x32,f16) x [sp;bl](32x16,f16) + unaries_tile(f32)
// Fragment mappings follow the CDNA5 16-bit A / 32-bit C VGPR layout tables.
// ---------------------------------------------------------------------------
__global__ void k_message(const float* __restrict__ SP,
                          const float* __restrict__ BL,
                          const _Float16* __restrict__ A,
                          const float* __restrict__ U,
                          float* __restrict__ Qo) {
  const int lane = threadIdx.x & 31;
  const int wave = threadIdx.x >> 5;
  const int tile = blockIdx.x * (blockDim.x >> 5) + wave;
  const int col  = lane & 15;          // matrix column: M-row for A, pixel for B/C/D
  const int hi   = lane >> 4;          // lane-group select
  const int pix  = tile * 16 + col;

  // A fragment: row M = col; VGPR vg holds K-pairs per the 16-bit A layout.
  v16h a;
  const _Float16* arow = A + col * 32;
  #pragma unroll
  for (int vg = 0; vg < 8; ++vg) {
    int kb = (vg < 4 ? 2 * vg : 16 + 2 * (vg - 4)) + hi * 8;
    a[2 * vg]     = arow[kb];
    a[2 * vg + 1] = arow[kb + 1];
  }

  // B fragment: column N = col (pixel); row K = channel (0..15 sp, 16..31 bl).
  const float* spp = SP + pix * 16;
  const float* blp = BL + pix * 16;
  v16h b;
  #pragma unroll
  for (int vg = 0; vg < 8; ++vg) {
    int kb = (vg < 4 ? 2 * vg : 16 + 2 * (vg - 4)) + hi * 8;
    #pragma unroll
    for (int pr = 0; pr < 2; ++pr) {
      int k = kb + pr;
      float v = (k < 16) ? spp[k] : blp[k - 16];
      b[2 * vg + pr] = (_Float16)v;
    }
  }

  // C fragment: unaries; element r -> row M = r + hi*8, column N = col.
  v8f c;
  const float* up = U + pix * 16;
  #pragma unroll
  for (int r = 0; r < 8; ++r) c[r] = up[r + hi * 8];

  v8f d = __builtin_amdgcn_wmma_f32_16x16x32_f16(
      /*neg_a=*/false, a, /*neg_b=*/false, b,
      /*c_mod=*/(short)0, c, /*reuse_a=*/false, /*reuse_b=*/false);

  float* qp = Qo + pix * 16;
  #pragma unroll
  for (int r = 0; r < 8; ++r) qp[r + hi * 8] = d[r];
}

// ---------------------------------------------------------------------------
// Host-side launcher (graph-capture safe: launches only).
// ---------------------------------------------------------------------------
extern "C" void kernel_launch(void* const* d_in, const int* in_sizes, int n_in,
                              void* d_out, int out_size, void* d_ws, size_t ws_size,
                              hipStream_t stream) {
  const float* img  = (const float*)d_in[0];
  const float* ubox = (const float*)d_in[1];
  const float* ugbl = (const float*)d_in[2];
  const float* Ws   = (const float*)d_in[3];
  const float* Wb   = (const float*)d_in[4];
  const float* Cm   = (const float*)d_in[5];
  const float* bw   = (const float*)d_in[6];
  const float* gw   = (const float*)d_in[7];

  float* U   = (float*)d_ws;
  float* Q   = U   + HW16;
  float* P   = Q   + HW16;
  float* SP  = P   + HW16;
  float* TB  = SP  + HW16;        // spatial tmp, then bilateral output
  float* W25 = TB  + HW16;        // 25 * HW
  float* RBN = W25 + 25 * HW;
  float* RSN = RBN + HW;
  float* GK  = RSN + HW;          // 19 floats (Gaussian taps)
  _Float16* A = (_Float16*)(GK + 32);

  // one-time precompute
  k_init_small<<<1, 576, 0, stream>>>(Ws, Wb, Cm, A, GK);
  k_prep<<<HW16 / 256, 256, 0, stream>>>(ubox, ugbl, bw, gw, U, Q);
  k_bw<<<HW / 256, 256, 0, stream>>>(img, GK, W25, RBN, RSN);

  // 5 mean-field iterations
  dim3 bgrid(Ww / TILE, Hh / TILE), bblock(TILE, TILE);
  for (int it = 0; it < 5; ++it) {
    k_softmax<<<HW / 256, 256, 0, stream>>>(Q, P);
    k_spat_v<<<HW * 4 / 256, 256, 0, stream>>>(P, TB, GK);
    k_spat_h<<<HW * 4 / 256, 256, 0, stream>>>(TB, SP, GK, RSN);
    k_bilateral<<<bgrid, bblock, 0, stream>>>(P, W25, RBN, TB);
    float* qout = (it == 4) ? (float*)d_out : Q;
    k_message<<<(HW / 16) / 8, 256, 0, stream>>>(SP, TB, A, U, qout);
  }
}
